// LSTM_89524298318146
// MI455X (gfx1250) — matrix operand order repaired and checked by reading, compiled, not run on hardware
//
#include <hip/hip_runtime.h>

// ---------------------------------------------------------------------------
// LSTM: B=2048, T=256, H=256, O=10
// Strategy: per-workgroup 16-row batch tile; h double-buffered in LDS (bf16);
// Wh pre-converted to bf16 WMMA-fragment layout in d_ws; 4H GEMM per step via
// v_wmma_f32_16x16x32_bf16; c state held in registers in D-matrix layout.
//
// Codegen details:
//  - weight-fragment base pointer is laundered through an empty asm each step
//    so LICM can't hoist the (t-invariant) weight loads (previously caused
//    ~512 VGPRs of hoisted fragments + scratch spills);
//  - the laundered pointer is kept in address_space(1) so the loads lower to
//    global_load_b128 (LOADcnt only), not flat_load_b128 (LOADcnt+DScnt);
//  - 16-byte vectors are clang ext_vector POD types (HIP's uint4 is a class
//    whose operator= cannot bind address_space(1) references);
//  - the x[B,T] slice for the block is preloaded transposed into LDS once,
//    removing a serialized global round-trip from every time step.
// ---------------------------------------------------------------------------

typedef __attribute__((ext_vector_type(16))) __bf16        v16bf;
typedef __attribute__((ext_vector_type(8)))  float         v8f;
typedef __attribute__((ext_vector_type(4)))  unsigned int  v4u;

typedef __attribute__((address_space(1))) const char  gchar;
typedef __attribute__((address_space(1))) const v4u   gv4u;

#define B_TOT 2048
#define T_LEN 256
#define H_DIM 256
#define O_DIM 10
#define BT    16          // batch rows per workgroup
#define HPAD  264         // padded LDS row stride in halfwords (bank-friendly)

__device__ __forceinline__ float sigf(float x) {
    return 1.0f / (1.0f + __expf(-x));
}
__device__ __forceinline__ float tanhfast(float x) {
    // tanh(x) = 2*sigmoid(2x) - 1
    return 2.0f / (1.0f + __expf(-2.0f * x)) - 1.0f;
}

// ---------------------------------------------------------------------------
// Prep: Wh [4][H][H] f32  ->  bf16 B-operand fragments for
// 16x16x32 bf16 WMMA, in contiguous k order (matching the A loads).
// frag index = ((((g*16 + nt)*8 + kt)*32 + lane)*16 + e)
//   B element (lane, e): N = nt*16 + (lane&15), K = kt*32 + e + 16*(lane>>4)
//   B[k][n] = Wh[g][n][k]   (so D[m][i] = sum_k h[m][k] * Wh[g][i][k])
// ---------------------------------------------------------------------------
__global__ __launch_bounds__(256) void lstm_prep_whfrag(
    const float* __restrict__ Wh, __bf16* __restrict__ frag) {
    int idx = blockIdx.x * 256 + threadIdx.x;
    if (idx >= 4 * 16 * 8 * 32 * 16) return;
    int e    = idx & 15;
    int lane = (idx >> 4) & 31;
    int kt   = (idx >> 9) & 7;
    int nt   = (idx >> 12) & 15;
    int g    = (idx >> 16) & 3;
    int i = nt * 16 + (lane & 15);
    int k = kt * 32 + e + 16 * (lane >> 4);
    frag[idx] = (__bf16)Wh[(g * H_DIM + i) * H_DIM + k];
}

// ---------------------------------------------------------------------------
// Main recurrent kernel. One workgroup = 16 batch rows, 8 waves.
// Wave w owns hidden columns [w*32, w*32+32) for all 4 gates.
// ---------------------------------------------------------------------------
__global__ __launch_bounds__(256) void lstm_step_kernel(
    const float* __restrict__ x,       // [B, T]
    const float* __restrict__ Wx,      // [4, H]
    const float* __restrict__ bx,      // [4, H]
    const float* __restrict__ bh,      // [4, H]
    const __bf16* __restrict__ whfrag, // [4*16*8*32*16] bf16 fragments
    const float* __restrict__ W_ph,    // [O, H]
    const float* __restrict__ b_ph,    // [O]
    float* __restrict__ out)           // [B, O]
{
    __shared__ __bf16 hbuf[2][BT * HPAD];     // ~16.9 KB
    __shared__ float  xallT[T_LEN][BT];       // 16 KB, x transposed [t][row]

    const int tid  = threadIdx.x;
    const int wave = tid >> 5;
    const int lane = tid & 31;
    const int row  = lane & 15;   // M / N index within a 16-wide tile
    const int hi   = lane >> 4;   // lane half-group
    const int bbase = blockIdx.x * BT;

    // zero h0 buffers
    for (int i = tid; i < BT * HPAD; i += 256) {
        hbuf[0][i] = (__bf16)0.0f;
        hbuf[1][i] = (__bf16)0.0f;
    }
    // preload entire x slice for this block, transposed (coalesced reads)
    for (int idx = tid; idx < BT * T_LEN; idx += 256) {
        const int r  = idx >> 8;    // row 0..15
        const int tt = idx & 255;   // time step
        xallT[tt][r] = x[(bbase + r) * T_LEN + tt];
    }

    // Preload per-lane Wx and (bx+bh) scalars for this wave's two column tiles
    float wxv[4][2], bsv[4][2];
    {
        int i0 = wave * 32 + row;
        int i1 = i0 + 16;
        #pragma unroll
        for (int g = 0; g < 4; ++g) {
            wxv[g][0] = Wx[g * H_DIM + i0];
            wxv[g][1] = Wx[g * H_DIM + i1];
            bsv[g][0] = bx[g * H_DIM + i0] + bh[g * H_DIM + i0];
            bsv[g][1] = bx[g * H_DIM + i1] + bh[g * H_DIM + i1];
        }
    }

    // c state in registers, laid out exactly as WMMA D: element e of tile tt
    // corresponds to (m = e + 8*hi, i = wave*32 + tt*16 + row)
    float cst[2][8];
    #pragma unroll
    for (int t = 0; t < 2; ++t)
        #pragma unroll
        for (int e = 0; e < 8; ++e) cst[t][e] = 0.0f;

    __syncthreads();

    for (int t = 0; t < T_LEN; ++t) {
        const int cur = t & 1;
        const int nxt = cur ^ 1;

        // Launder the weight base pointer (kept in address_space(1)) so the
        // t-invariant fragment loads stay inside the loop as global_loads.
        gchar* gb = (gchar*)whfrag;
        asm volatile("" : "+s"(gb));
        gchar* fbase = gb + lane * 32;

        // ---- load A fragments (h_t) from LDS: 16x32 bf16 tiles, kt = 0..7
        v16bf a[8];
        {
            const __bf16* hb = &hbuf[cur][row * HPAD + 16 * hi];
            #pragma unroll
            for (int kt = 0; kt < 8; ++kt) {
                union { v4u q[2]; v16bf v; } au;
                au.q[0] = *(const v4u*)(hb + kt * 32);
                au.q[1] = *(const v4u*)(hb + kt * 32 + 8);
                a[kt] = au.v;
            }
        }

        // ---- per-lane xt values for the 8 M rows of this lane's half-group
        float xv[8];
        {
            const float* xp = &xallT[t][8 * hi];
            #pragma unroll
            for (int e = 0; e < 8; ++e) xv[e] = xp[e];
        }

        // ---- init accumulators: z = xt*Wx + (bx+bh); j = g*2 + tt
        v8f z[8];
        #pragma unroll
        for (int j = 0; j < 8; ++j) {
            const int g = j >> 1, tt = j & 1;
            #pragma unroll
            for (int e = 0; e < 8; ++e)
                z[j][e] = xv[e] * wxv[g][tt] + bsv[g][tt];
        }

        // ---- 64 WMMAs: kt outer, 8 independent accumulator chains inner
        #pragma unroll
        for (int kt = 0; kt < 8; ++kt) {
            #pragma unroll
            for (int j = 0; j < 8; ++j) {
                const int g = j >> 1, tt = j & 1;
                const int nt = wave * 2 + tt; // N-tile index within gate
                const int blk = (((g * 16 + nt) * 8 + kt) << 10); // bytes
                union { v4u q[2]; v16bf v; } bu;
                bu.q[0] = *(gv4u*)(fbase + blk);
                bu.q[1] = *(gv4u*)(fbase + blk + 16);
                z[j] = __builtin_amdgcn_wmma_f32_16x16x32_bf16(
                    false, a[kt], false, bu.v, (short)0, z[j], false, false);
            }
        }

        // ---- gate nonlinearities + state update (pure per-lane VALU)
        #pragma unroll
        for (int tt = 0; tt < 2; ++tt) {
            const int col = wave * 32 + tt * 16 + row;
            #pragma unroll
            for (int e = 0; e < 8; ++e) {
                float zg = tanhfast(z[0 * 2 + tt][e]);
                float zi = sigf(z[1 * 2 + tt][e]);
                float zf = sigf(z[2 * 2 + tt][e]);
                float zo = sigf(z[3 * 2 + tt][e]);
                float c  = zg * zi + cst[tt][e] * zf;
                cst[tt][e] = c;
                float h = tanhfast(c) * zo;
                const int m = e + 8 * hi;
                hbuf[nxt][m * HPAD + col] = (__bf16)h;
            }
        }
        __syncthreads();
    }

    // ---- output projection: out[b, o] = h_T[b] . W_ph[o] + b_ph[o]
    // final h is in hbuf[T_LEN & 1] (T even -> hbuf[0])
    const __bf16* hf = hbuf[T_LEN & 1];
    if (tid < BT * O_DIM) {
        const int m = tid / O_DIM;
        const int o = tid % O_DIM;
        float s = b_ph[o];
        const float* wp = &W_ph[o * H_DIM];
        #pragma unroll 8
        for (int k = 0; k < H_DIM; ++k)
            s += (float)hf[m * HPAD + k] * wp[k];
        out[(bbase + m) * O_DIM + o] = s;
    }
}

// ---------------------------------------------------------------------------
// Launch
// ---------------------------------------------------------------------------
extern "C" void kernel_launch(void* const* d_in, const int* in_sizes, int n_in,
                              void* d_out, int out_size, void* d_ws, size_t ws_size,
                              hipStream_t stream) {
    (void)in_sizes; (void)n_in; (void)out_size; (void)ws_size;
    const float* x    = (const float*)d_in[0]; // [B, T]
    const float* Wx   = (const float*)d_in[1]; // [4, H]
    const float* bx   = (const float*)d_in[2]; // [4, H]
    const float* Wh   = (const float*)d_in[3]; // [4, H, H]
    const float* bh   = (const float*)d_in[4]; // [4, H]
    const float* W_ph = (const float*)d_in[5]; // [O, H]
    const float* b_ph = (const float*)d_in[6]; // [O]
    float* out = (float*)d_out;

    __bf16* whfrag = (__bf16*)d_ws; // 4*16*8*32*16 * 2B = 512 KB

    // Convert Wh -> bf16 fragment layout (262144 elements)
    lstm_prep_whfrag<<<1024, 256, 0, stream>>>(Wh, whfrag);

    // Recurrent kernel: 2048/16 = 128 workgroups of 256 threads (8 waves)
    lstm_step_kernel<<<B_TOT / BT, 256, 0, stream>>>(
        x, Wx, bx, bh, whfrag, W_ph, b_ph, out);
}
